// MyViT_30219389894889
// MI455X (gfx1250) — matrix operand order repaired and checked by reading, not compiled, not run on hardware
//
#include <hip/hip_runtime.h>
#include <hip/hip_bf16.h>
#include <math.h>

// ---------------- constants ----------------
#define BATCH 128
#define SEQL  197
#define SPAD  224          // 14 * 16 padded sequence
#define HID   768
#define NHEAD 12
#define DHEAD 64
#define NPATCH 196
#define LN_EPS 1e-5f

#define MT_IMG 13          // ceil(196/16)
#define KT_IMG 24          // 768/32
#define MT_SEQ 14          // 224/16
#define TILE_ELEMS 512     // 32 lanes * 16 bf16 per operand tile

typedef __attribute__((ext_vector_type(16))) __bf16 v16bf;
typedef __attribute__((ext_vector_type(8)))  float  v8f;
typedef __attribute__((ext_vector_type(4)))  unsigned v4u;
typedef __attribute__((ext_vector_type(8)))  unsigned v8u;

// ---------------- WMMA helpers ----------------
__device__ __forceinline__ v8f wmma_bf16(v16bf a, v16bf b, v8f c) {
    return __builtin_amdgcn_wmma_f32_16x16x32_bf16(
        false, a, false, b, (short)0, c, false, false);
}

// A-matrix (16-bit 16x32) per-lane K index for element e (ISA 7.12.2)
__device__ __forceinline__ int amap_k(int e, int half) {
    int v = e >> 1, lo = e & 1;
    int kb = half ? 8 : 0;
    return (v < 4) ? (kb + 2 * v + lo) : (16 + kb + 2 * (v - 4) + lo);
}
// B-matrix (32x16): same K order as A (lanes 0-15: K=0..7,16..23; lanes 16-31: K=8..15,24..31)
__device__ __forceinline__ int bmap_k(int e, int half) {
    return amap_k(e, half);
}

// Packed operand tile loads: lane's 16 bf16 contiguous -> 2x b128 load
__device__ __forceinline__ v16bf ldp(const __bf16* p) {
    return *(const v16bf*)p;
}
// Non-temporal variant for single-use streaming operands
__device__ __forceinline__ v16bf ldp_nt(const __bf16* p) {
    return __builtin_nontemporal_load((const v16bf*)p);
}

// ---------------- Tensor Data Mover: 1-D bulk copy global -> LDS ------------
// Builds a D# (group0 + group1) in SGPRs and issues TENSOR_LOAD_TO_LDS.
// nbytes must be < 65536 and a multiple of 4. Call from a single wave;
// includes s_wait_tensorcnt; follow with a workgroup barrier.
__device__ __forceinline__ void tdm_load_1d(const void* gptr, unsigned lds_off,
                                            unsigned nbytes) {
    unsigned long long ga = (unsigned long long)gptr;
    v4u g0;
    g0[0] = 1u;                                   // count=1, user descriptor
    g0[1] = lds_off;                              // lds_addr (bytes)
    g0[2] = (unsigned)ga;                         // global_addr[31:0]
    g0[3] = (unsigned)(ga >> 32) | (2u << 30);    // global_addr[56:32] | type=2
    v8u g1;
    g1[0] = 0u;                                   // wg_mask=0, data_size=1B
    g1[1] = (nbytes & 0xFFFFu) << 16;             // tensor_dim0[15:0]
    g1[2] = (nbytes >> 16) | (1u << 16);          // tensor_dim0[31:16], dim1=1
    g1[3] = (nbytes & 0xFFFFu) << 16;             // tile_dim0
    g1[4] = 0u;                                   // tile_dim1=0, tile_dim2=0
    g1[5] = nbytes;                               // tensor_dim0_stride lo
    g1[6] = 0u;
    g1[7] = 0u;
    asm volatile("tensor_load_to_lds %0, %1" ::"s"(g0), "s"(g1) : "memory");
    __builtin_amdgcn_s_wait_tensorcnt(0);
}

// ---------------- kernel 0: pack weights into B-operand tiles ----------------
__global__ void prep_weights_kernel(const float* __restrict__ wmap,
                                    const float* __restrict__ wq,
                                    const float* __restrict__ wk,
                                    const float* __restrict__ wv,
                                    __bf16* __restrict__ wtp,
                                    __bf16* __restrict__ wqp,
                                    __bf16* __restrict__ wkp,
                                    __bf16* __restrict__ wvp) {
    const int NW = KT_IMG * 48 * TILE_ELEMS;          // 589824
    const int NH = NHEAD * 8 * TILE_ELEMS;            // 49152
    int idx = blockIdx.x * blockDim.x + threadIdx.x;
    if (idx < NW) {
        int e = idx & 15, lane = (idx >> 4) & 31, tile = idx >> 9;
        int kt = tile / 48, nt = tile % 48;
        int kk = kt * 32 + bmap_k(e, lane >> 4);
        int n = nt * 16 + (lane & 15);
        wtp[idx] = (__bf16)wmap[(long)n * HID + kk];     // B[k][n] = W_map[n][k]
    } else if (idx < NW + 3 * NH) {
        int r = idx - NW;
        int mat = r / NH, rr = r % NH;
        int e = rr & 15, lane = (rr >> 4) & 31, tile = rr >> 9;
        int h = tile >> 3, t8 = tile & 7;
        int kt = t8 >> 2, nt = t8 & 3;
        int kk = kt * 32 + bmap_k(e, lane >> 4);
        int n = nt * 16 + (lane & 15);
        const float* src = (mat == 0) ? wq : (mat == 1) ? wk : wv;
        __bf16* dst = (mat == 0) ? wqp : (mat == 1) ? wkp : wvp;
        dst[rr] = (__bf16)src[(long)h * DHEAD * DHEAD + (long)n * DHEAD + kk];
    }
}

// ---------------- kernel 1: pack images into A-operand tiles ----------------
__global__ void img_pack_kernel(const float* __restrict__ images,
                                __bf16* __restrict__ ip) {
    long idx = (long)blockIdx.x * blockDim.x + threadIdx.x;
    const long total = (long)BATCH * MT_IMG * KT_IMG * TILE_ELEMS;
    if (idx >= total) return;
    int e = (int)(idx & 15);
    int lane = (int)((idx >> 4) & 31);
    long t = idx >> 9;
    int tile = (int)(t % (MT_IMG * KT_IMG));
    int b = (int)(t / (MT_IMG * KT_IMG));
    int mt = tile / KT_IMG, kt = tile % KT_IMG;
    int m = mt * 16 + (lane & 15);
    float val = 0.0f;
    if (m < NPATCH) {
        int k = kt * 32 + amap_k(e, lane >> 4);
        val = images[(long)b * NPATCH * HID + (long)m * HID + k];
    }
    ip[idx] = (__bf16)val;
}

// ---------------- kernel 2: patch-embed GEMM (TDM A-strip + WMMA) -----------
__global__ __launch_bounds__(256) void embed_gemm_kernel(
    const __bf16* __restrict__ ip, const __bf16* __restrict__ wtp,
    float* __restrict__ out) {
    int mt = blockIdx.x;   // 0..12
    int b = blockIdx.y;
    const __bf16* A = ip + ((long)b * MT_IMG + mt) * KT_IMG * TILE_ELEMS;

    // DMA the whole 24 KB A strip into LDS once per block (wave 0 issues).
    __shared__ __bf16 lds_a[KT_IMG * TILE_ELEMS];
    if (threadIdx.x < 32)
        tdm_load_1d(A, (unsigned)(unsigned long long)(uintptr_t)&lds_a[0],
                    KT_IMG * TILE_ELEMS * 2);
    __syncthreads();

    int wave = threadIdx.x >> 5, lane = threadIdx.x & 31;
    v8f acc[6] = {};
    for (int kk = 0; kk < KT_IMG; ++kk) {
        v16bf a = ldp(lds_a + kk * TILE_ELEMS + lane * 16);  // ds_load_b128 x2
        // preload all 6 B tiles so loads clause together and overlap the
        // WMMA chain instead of serializing wait->wmma->load
        v16bf bt[6];
#pragma unroll
        for (int j = 0; j < 6; ++j)
            bt[j] = ldp(wtp + ((long)kk * 48 + wave * 6 + j) * TILE_ELEMS + lane * 16);
#pragma unroll
        for (int j = 0; j < 6; ++j) acc[j] = wmma_bf16(a, bt[j], acc[j]);
    }
    int n = lane & 15, mh = lane >> 4;
#pragma unroll
    for (int j = 0; j < 6; ++j) {
        int nt = wave * 6 + j;
#pragma unroll
        for (int r = 0; r < 8; ++r) {
            int m = mt * 16 + r + 8 * mh;
            if (m < NPATCH)
                out[((long)b * SEQL + (m + 1)) * HID + nt * 16 + n] = acc[j][r];
        }
    }
}

// ---------------- kernel 3: class token + bias + positional emb -------------
__global__ void posbias_kernel(const float* __restrict__ cls,
                               const float* __restrict__ bmap,
                               float* __restrict__ out) {
    long idx = (long)blockIdx.x * blockDim.x + threadIdx.x;
    const long total = (long)BATCH * SEQL * HID;
    if (idx >= total) return;
    int d = (int)(idx % HID);
    int s = (int)((idx / HID) % SEQL);
    float jf = (float)(d & ~1);
    float scale = __expf(-(jf / (float)HID) * logf(10000.0f));
    float ang = (float)s * scale;
    float pe = (d & 1) ? __cosf(ang) : __sinf(ang);
    if (s == 0)
        out[idx] = cls[d] + pe;
    else
        out[idx] += bmap[d] + pe;
}

// ---------------- kernel 4: per-sample LN statistics ------------------------
__global__ __launch_bounds__(256) void ln_stats_kernel(const float* __restrict__ out,
                                                       float* __restrict__ stats) {
    int b = blockIdx.x;
    const float* p = out + (long)b * SEQL * HID;
    const int N = SEQL * HID;
    float s = 0.f, s2 = 0.f;
    for (int i = threadIdx.x; i < N; i += blockDim.x) {
        float v = p[i];
        s += v;
        s2 += v * v;
    }
    __shared__ float sh[256], sh2[256];
    sh[threadIdx.x] = s;
    sh2[threadIdx.x] = s2;
    __syncthreads();
    for (int o = 128; o > 0; o >>= 1) {
        if (threadIdx.x < o) {
            sh[threadIdx.x] += sh[threadIdx.x + o];
            sh2[threadIdx.x] += sh2[threadIdx.x + o];
        }
        __syncthreads();
    }
    if (threadIdx.x == 0) {
        float mean = sh[0] / (float)N;
        float var = sh2[0] / (float)N - mean * mean;
        stats[2 * b] = mean;
        stats[2 * b + 1] = rsqrtf(var + LN_EPS);
    }
}

// ---------------- kernel 5: LN apply -> per-head A-operand tiles ------------
__global__ void ln_pack_kernel(const float* __restrict__ out,
                               const float* __restrict__ stats,
                               const float* __restrict__ lnw,
                               const float* __restrict__ lnb,
                               __bf16* __restrict__ xp) {
    long idx = (long)blockIdx.x * blockDim.x + threadIdx.x;
    const long total = (long)BATCH * NHEAD * 28 * TILE_ELEMS;
    if (idx >= total) return;
    int e = (int)(idx & 15);
    int lane = (int)((idx >> 4) & 31);
    long t = idx >> 9;
    int tile = (int)(t % 28);
    int h = (int)((t / 28) % NHEAD);
    int b = (int)(t / (28 * NHEAD));
    int mt = tile >> 1, kt = tile & 1;
    int s = mt * 16 + (lane & 15);
    float val = 0.0f;
    if (s < SEQL) {
        int d = h * DHEAD + kt * 32 + amap_k(e, lane >> 4);
        long sd = (long)s * HID + d;
        float mean = stats[2 * b], rstd = stats[2 * b + 1];
        val = (out[(long)b * SEQL * HID + sd] - mean) * rstd * lnw[sd] + lnb[sd];
    }
    xp[idx] = (__bf16)val;
}

// ---------------- kernel 6: Q/K/V projection + operand repack ---------------
// grid (BATCH*NHEAD, 3); blockIdx.y selects matrix (0:q 1:k 2:v)
__global__ __launch_bounds__(256) void qkv_pack_kernel(
    const __bf16* __restrict__ xp, const __bf16* __restrict__ wqp,
    const __bf16* __restrict__ wkp, const __bf16* __restrict__ wvp,
    const float* __restrict__ bq, const float* __restrict__ bk,
    const float* __restrict__ bv, __bf16* __restrict__ qp,
    __bf16* __restrict__ kp, __bf16* __restrict__ vp) {
    int mat = blockIdx.y;
    int h = blockIdx.x % NHEAD, b = blockIdx.x / NHEAD;
    long hb = (long)(b * NHEAD + h);
    const __bf16* wp =
        ((mat == 0) ? wqp : (mat == 1) ? wkp : wvp) + (long)h * 8 * TILE_ELEMS;
    const float* bias = ((mat == 0) ? bq : (mat == 1) ? bk : bv) + h * DHEAD;
    const __bf16* xpb = xp + hb * 28 * TILE_ELEMS;

    // DMA the per-(b,h) 28 KB packed-x strip into LDS (wave 0 issues).
    __shared__ __bf16 xa[28 * TILE_ELEMS];          // 28672 B
    __shared__ __bf16 m_lds[SPAD][DHEAD + 8];       // 32256 B
    if (threadIdx.x < 32)
        tdm_load_1d(xpb, (unsigned)(unsigned long long)(uintptr_t)&xa[0],
                    28 * TILE_ELEMS * 2);
    __syncthreads();

    int wave = threadIdx.x >> 5, lane = threadIdx.x & 31;
    int n = lane & 15, mh = lane >> 4;

    // compute 14x4 output tiles into LDS
    for (int t = wave; t < 56; t += 8) {
        int mt = t / 4, nt = t % 4;
        v16bf a0 = ldp(xa + (mt * 2 + 0) * TILE_ELEMS + lane * 16);
        v16bf a1 = ldp(xa + (mt * 2 + 1) * TILE_ELEMS + lane * 16);
        v16bf b0 = ldp(wp + (long)(0 * 4 + nt) * TILE_ELEMS + lane * 16);
        v16bf b1 = ldp(wp + (long)(1 * 4 + nt) * TILE_ELEMS + lane * 16);
        v8f acc = {};
        acc = wmma_bf16(a0, b0, acc);
        acc = wmma_bf16(a1, b1, acc);
        float bia = bias[nt * 16 + n];
#pragma unroll
        for (int r = 0; r < 8; ++r)
            m_lds[mt * 16 + r + 8 * mh][nt * 16 + n] = (__bf16)(acc[r] + bia);
    }
    __syncthreads();

    // repack LDS into packed operand tiles (28 tiles x 32 lane-blocks)
    __bf16* dstbase = ((mat == 0) ? qp : (mat == 1) ? kp : vp) + hb * 28 * TILE_ELEMS;
    for (int it = threadIdx.x; it < 28 * 32; it += 256) {
        int tile = it >> 5, vl = it & 31;
        int vh = vl >> 4, vn = vl & 15;
        v16bf o;
        if (mat == 0) {            // q: A-format [mt][kt]
            int mtl = tile >> 1, kt = tile & 1;
            int row = mtl * 16 + vn;
#pragma unroll
            for (int e = 0; e < 16; ++e) o[e] = m_lds[row][kt * 32 + amap_k(e, vh)];
        } else if (mat == 1) {     // k^T: B-format [kt][tt], B[d][t] = k[t][d]
            int kt = tile / MT_SEQ, tt = tile % MT_SEQ;
            int trow = tt * 16 + vn;
#pragma unroll
            for (int e = 0; e < 16; ++e) o[e] = m_lds[trow][kt * 32 + bmap_k(e, vh)];
        } else {                   // v: B-format [kt=7][nt=4]
            int kt = tile >> 2, nt = tile & 3;
#pragma unroll
            for (int e = 0; e < 16; ++e)
                o[e] = m_lds[kt * 32 + bmap_k(e, vh)][nt * 16 + vn];
        }
        *(v16bf*)(dstbase + (long)tile * TILE_ELEMS + vl * 16) = o;
    }
}

// ---------------- kernel 7: attention (scores -> softmax -> att@V) ----------
__global__ __launch_bounds__(256) void attn_kernel(const __bf16* __restrict__ qp,
                                                   const __bf16* __restrict__ kp,
                                                   const __bf16* __restrict__ vp,
                                                   float* __restrict__ out) {
    int sq = blockIdx.x;  // 64-row query strip
    int h = blockIdx.y;
    int b = blockIdx.z;
    long hb = (long)(b * NHEAD + h);
    const __bf16* qb = qp + hb * 28 * TILE_ELEMS;
    const __bf16* kb = kp + hb * 28 * TILE_ELEMS;
    const __bf16* vb = vp + hb * 28 * TILE_ELEMS;

    __shared__ float sc[64][SPAD + 8];  // 59392 B
    int wave = threadIdx.x >> 5, lane = threadIdx.x & 31;
    int n = lane & 15, mh = lane >> 4;

    // Phase 1: scores = (q @ k^T)/8, mask t >= SEQL. 4 local x 14 tiles.
    for (int t = wave; t < 56; t += 8) {
        int mt = t / MT_SEQ, nt = t % MT_SEQ;
        int mt_g = sq * 4 + mt;
        if (mt_g < MT_SEQ) {
            v16bf a0 = ldp_nt(qb + (long)(mt_g * 2 + 0) * TILE_ELEMS + lane * 16);
            v16bf a1 = ldp_nt(qb + (long)(mt_g * 2 + 1) * TILE_ELEMS + lane * 16);
            v16bf b0 = ldp_nt(kb + (long)(0 * MT_SEQ + nt) * TILE_ELEMS + lane * 16);
            v16bf b1 = ldp_nt(kb + (long)(1 * MT_SEQ + nt) * TILE_ELEMS + lane * 16);
            v8f acc = {};
            acc = wmma_bf16(a0, b0, acc);
            acc = wmma_bf16(a1, b1, acc);
#pragma unroll
            for (int r = 0; r < 8; ++r) {
                int m = mt * 16 + r + 8 * mh;
                int tt = nt * 16 + n;
                float val = acc[r] * 0.125f;
                if (tt >= SEQL) val = -1e30f;
                sc[m][tt] = val;
            }
        } else {
#pragma unroll
            for (int r = 0; r < 8; ++r)
                sc[mt * 16 + r + 8 * mh][nt * 16 + n] = -1e30f;
        }
    }
    __syncthreads();

    // Phase 2: softmax, 4 threads per row via shuffles
    {
        int row = threadIdx.x >> 2;
        int part = threadIdx.x & 3;
        float mx = -1e30f;
        for (int t = part; t < SPAD; t += 4) mx = fmaxf(mx, sc[row][t]);
        mx = fmaxf(mx, __shfl_xor(mx, 1, 4));
        mx = fmaxf(mx, __shfl_xor(mx, 2, 4));
        float sum = 0.f;
        for (int t = part; t < SPAD; t += 4) {
            float e = __expf(sc[row][t] - mx);
            sc[row][t] = e;
            sum += e;
        }
        sum += __shfl_xor(sum, 1, 4);
        sum += __shfl_xor(sum, 2, 4);
        float inv = 1.0f / sum;
        for (int t = part; t < SPAD; t += 4) sc[row][t] *= inv;
    }
    __syncthreads();

    // Phase 3: msa = att @ v accumulated onto tokens. 4x4 tiles.
    for (int t = wave; t < 16; t += 8) {
        int mt = t >> 2, nt = t & 3;
        if (sq * 64 + mt * 16 >= SEQL) continue;
        v8f acc = {};
#pragma unroll
        for (int kk = 0; kk < SPAD / 32; ++kk) {
            v16bf a;
            {
                int r = mt * 16 + (lane & 15);
                const float* p = &sc[r][kk * 32];
#pragma unroll
                for (int e = 0; e < 16; ++e) a[e] = (__bf16)p[amap_k(e, mh)];
            }
            v16bf bb = ldp_nt(vb + (long)(kk * 4 + nt) * TILE_ELEMS + lane * 16);
            acc = wmma_bf16(a, bb, acc);
        }
#pragma unroll
        for (int r = 0; r < 8; ++r) {
            int s = sq * 64 + mt * 16 + r + 8 * mh;
            if (s < SEQL) {
                long o = ((long)b * SEQL + s) * HID + h * DHEAD + nt * 16 + n;
                out[o] += acc[r];
            }
        }
    }
}

// ---------------- launcher ----------------
extern "C" void kernel_launch(void* const* d_in, const int* in_sizes, int n_in,
                              void* d_out, int out_size, void* d_ws, size_t ws_size,
                              hipStream_t stream) {
    (void)in_sizes; (void)n_in; (void)out_size; (void)ws_size;
    const float* images = (const float*)d_in[0];
    const float* W_map  = (const float*)d_in[1];
    const float* b_map  = (const float*)d_in[2];
    const float* cls    = (const float*)d_in[3];
    const float* ln_w   = (const float*)d_in[4];
    const float* ln_b   = (const float*)d_in[5];
    const float* Wq     = (const float*)d_in[6];
    const float* bq     = (const float*)d_in[7];
    const float* Wk     = (const float*)d_in[8];
    const float* bk     = (const float*)d_in[9];
    const float* Wv     = (const float*)d_in[10];
    const float* bv     = (const float*)d_in[11];
    float* out = (float*)d_out;

    char* ws = (char*)d_ws;
    size_t off = 0;
    auto carve = [&](size_t bytes) -> void* {
        void* p = ws + off;
        off = (off + bytes + 255) & ~(size_t)255;
        return p;
    };
    const size_t NWT = (size_t)KT_IMG * 48 * TILE_ELEMS;          // 589824
    const size_t NHT = (size_t)NHEAD * 8 * TILE_ELEMS;            // 49152
    const size_t NIP = (size_t)BATCH * MT_IMG * KT_IMG * TILE_ELEMS;
    const size_t NXP = (size_t)BATCH * NHEAD * 28 * TILE_ELEMS;

    __bf16* wtp = (__bf16*)carve(NWT * 2);
    __bf16* wqp = (__bf16*)carve(NHT * 2);
    __bf16* wkp = (__bf16*)carve(NHT * 2);
    __bf16* wvp = (__bf16*)carve(NHT * 2);
    __bf16* ip  = (__bf16*)carve(NIP * 2);
    __bf16* xpb = (__bf16*)carve(NXP * 2);
    __bf16* qpb = (__bf16*)carve(NXP * 2);
    __bf16* kpb = (__bf16*)carve(NXP * 2);
    __bf16* vpb = (__bf16*)carve(NXP * 2);
    float* stats = (float*)carve((size_t)BATCH * 2 * 4);

    {
        int total = (int)(NWT + 3 * NHT);  // 737280
        prep_weights_kernel<<<(total + 255) / 256, 256, 0, stream>>>(
            W_map, Wq, Wk, Wv, wtp, wqp, wkp, wvp);
    }
    img_pack_kernel<<<(unsigned)((NIP + 255) / 256), 256, 0, stream>>>(images, ip);
    embed_gemm_kernel<<<dim3(MT_IMG, BATCH), 256, 0, stream>>>(ip, wtp, out);
    {
        long total = (long)BATCH * SEQL * HID;
        posbias_kernel<<<(unsigned)((total + 255) / 256), 256, 0, stream>>>(cls, b_map, out);
    }
    ln_stats_kernel<<<BATCH, 256, 0, stream>>>(out, stats);
    ln_pack_kernel<<<(unsigned)((NXP + 255) / 256), 256, 0, stream>>>(out, stats,
                                                                      ln_w, ln_b, xpb);
    qkv_pack_kernel<<<dim3(BATCH * NHEAD, 3), 256, 0, stream>>>(
        xpb, wqp, wkp, wvp, bq, bk, bv, qpb, kpb, vpb);
    attn_kernel<<<dim3(4, NHEAD, BATCH), 256, 0, stream>>>(qpb, kpb, vpb, out);
}
